// CustomAttention_10539849744720
// MI455X (gfx1250) — matrix operand order repaired and checked by reading, not compiled, not run on hardware
//
#include <hip/hip_runtime.h>
#include <hip/hip_bf16.h>

// ---------------------------------------------------------------------------
// MHA (RoPE + causal flash attention) for MI455X / gfx1250, wave32 + WMMA f16,
// TDM (tensor_load_to_lds) double-buffered GEMM staging.
// B=2, S=2048, D=4096, H=32, HD=128.
// ---------------------------------------------------------------------------

typedef _Float16 h16;
typedef __attribute__((ext_vector_type(4)))  _Float16 v4h;
typedef __attribute__((ext_vector_type(8)))  _Float16 v8h;
typedef __attribute__((ext_vector_type(16))) _Float16 v16h;
typedef __attribute__((ext_vector_type(4)))  float    v4f;
typedef __attribute__((ext_vector_type(8)))  float    v8f;
typedef __attribute__((ext_vector_type(4)))  unsigned int u32x4;
typedef __attribute__((ext_vector_type(4)))  int      i32x4;
typedef __attribute__((ext_vector_type(8)))  int      i32x8;

#define DEV __device__ __forceinline__

// ---- WMMA fragment helpers (wave32, 16x16x32 f16, f32 accum) --------------
// A-matrix 16x32: lane = (m, half); element e -> K = 16*(e/8) + 8*half + e%8
DEV v16h ld_afrag(const h16* rowp, int halfsel) {
    v8h lo = *(const v8h*)(rowp + halfsel * 8);
    v8h hi = *(const v8h*)(rowp + 16 + halfsel * 8);
    v16h r;
#pragma unroll
    for (int i = 0; i < 8; ++i) { r[i] = lo[i]; r[i + 8] = hi[i]; }
    return r;
}

// B-matrix 32x16: lane = (n, half); element e -> K = 16*(e/8) + 8*half + e%8
DEV v16h ld_bfrag(const h16* colp, int halfsel) {
    v8h lo = *(const v8h*)(colp + halfsel * 8);
    v8h hi = *(const v8h*)(colp + 16 + halfsel * 8);
    v16h r;
#pragma unroll
    for (int i = 0; i < 8; ++i) { r[i] = lo[i]; r[i + 8] = hi[i]; }
    return r;
}

DEV v8f wmma16(v16h a, v16h b, v8f c) {
    return __builtin_amdgcn_wmma_f32_16x16x32_f16(
        /*neg_a=*/false, a, /*neg_b=*/false, b,
        /*c_mod=*/(short)0, c, /*reuse_a=*/false, /*reuse_b=*/false);
}

// ---------------------------------------------------------------------------
// TDM: issue a 2D tile load (Global -> LDS) via the Tensor Data Mover.
//   - 2-byte elements, tile = tile_k x tile_rows, row stride = stride_elems
//   - LDS padding: 4 DWORDs appended after every 32 DWORDs (tile_k must be 64)
//     -> LDS row stride = 72 halves
// Issue from ONE wave only (TDM ignores EXEC; one DMA per issuing wave).
// Tracked with TENSORcnt.
// ---------------------------------------------------------------------------
DEV void tdm_load_2d_64(unsigned lds_byte_off, const void* gptr,
                        unsigned tile_rows, unsigned stride_elems,
                        unsigned tensor_rows) {
    unsigned long long ga = (unsigned long long)(__SIZE_TYPE__)gptr;
    u32x4 g0;
    g0[0] = 1u;                                        // count=1, user mode
    g0[1] = lds_byte_off;                              // LDS dest (bytes)
    g0[2] = (unsigned)ga;                              // global addr [31:0]
    g0[3] = (unsigned)((ga >> 32) & 0x01ffffffull)     // global addr [56:32]
          | (2u << 30);                                // type = 2 ("image")
    const unsigned td0 = stride_elems;                 // tensor dim0 (K extent)
    const unsigned td1 = tensor_rows;                  // tensor dim1 (rows)
    i32x8 g1;
    g1[0] = (int)((1u << 16)      // data_size = 2 bytes
                | (1u << 20)      // pad_enable
                | (4u << 22)      // pad_interval: 32 DWORDs
                | (3u << 25));    // pad_amount:   4 DWORDs
    g1[1] = (int)((td0 & 0xffffu) << 16);                            // td0 lo
    g1[2] = (int)(((td0 >> 16) & 0xffffu) | ((td1 & 0xffffu) << 16));// td0 hi|td1 lo
    g1[3] = (int)(((td1 >> 16) & 0xffffu) | (64u << 16));            // td1 hi|tile_dim0=64
    g1[4] = (int)(tile_rows & 0xffffu);                              // tile_dim1, tile_dim2=0
    g1[5] = (int)stride_elems;                                       // dim0 stride lo32
    g1[6] = 0;
    g1[7] = 0;
    i32x4 z4; z4[0] = 0; z4[1] = 0; z4[2] = 0; z4[3] = 0;
#if defined(__clang_major__) && (__clang_major__ >= 23)
    i32x8 z8;
#pragma unroll
    for (int i = 0; i < 8; ++i) z8[i] = 0;
    __builtin_amdgcn_tensor_load_to_lds(g0, g1, z4, z4, z8, 0);
#else
    __builtin_amdgcn_tensor_load_to_lds(g0, g1, z4, z4, 0);
#endif
}

DEV unsigned lds_offset_of(const void* p) {
    return (unsigned)(unsigned long long)(__SIZE_TYPE__)p;
}

// ---------------------------------------------------------------------------
// fp32 -> f16 conversion (vectorized)
// ---------------------------------------------------------------------------
__global__ __launch_bounds__(256) void cvt_f32_f16(const float* __restrict__ src,
                                                   h16* __restrict__ dst, int n) {
    int i = (blockIdx.x * 256 + threadIdx.x) * 4;
    if (i + 3 < n) {
        v4f v = *(const v4f*)(src + i);
        v4h o;
#pragma unroll
        for (int j = 0; j < 4; ++j) o[j] = (h16)v[j];
        *(v4h*)(dst + i) = o;
    }
}

// ---------------------------------------------------------------------------
// WMMA GEMM with TDM staging: C[M,N] = A[M,K] @ B[N,K]^T (nn.Linear form).
// Block = 256 threads = 8 waves; 128x128 tile; each wave -> 16x128 strip.
// K consumed in 64-wide slabs, double-buffered in LDS via tensor_load_to_lds.
// ---------------------------------------------------------------------------
template <bool STORE_F32>
__global__ __launch_bounds__(256) void wmma_gemm_tdm(const h16* __restrict__ A,
                                                     const h16* __restrict__ B,
                                                     void* __restrict__ Cout,
                                                     int M, int N, int K) {
    constexpr int KS  = 64;   // K slab (elements)
    constexpr int STR = 72;   // LDS halves per row (64 + 4-DWORD TDM pad)
    __shared__ h16 Alds[2][128 * STR];
    __shared__ h16 Blds[2][128 * STR];

    const int t = threadIdx.x, wave = t >> 5, lane = t & 31;
    const int nlane = lane & 15, mbase = (lane >> 4) * 8, halfsel = lane >> 4;
    const int m0 = blockIdx.x * 128, n0 = blockIdx.y * 128;
    const int NS = K / KS;

    const h16* aTile = A + (size_t)m0 * K;
    const h16* bTile = B + (size_t)n0 * K;

    v8f acc[8];
#pragma unroll
    for (int j = 0; j < 8; ++j)
#pragma unroll
        for (int i = 0; i < 8; ++i) acc[j][i] = 0.0f;

    // Prologue: prefetch slabs 0 and 1 (A+B each -> 2 TDM ops per slab).
    if (wave == 0) {
        tdm_load_2d_64(lds_offset_of(&Alds[0][0]), aTile,       128, K, M);
        tdm_load_2d_64(lds_offset_of(&Blds[0][0]), bTile,       128, K, N);
        tdm_load_2d_64(lds_offset_of(&Alds[1][0]), aTile + KS,  128, K, M);
        tdm_load_2d_64(lds_offset_of(&Blds[1][0]), bTile + KS,  128, K, N);
    }

    for (int i = 0; i < NS; ++i) {
        if (wave == 0) {
            if (i + 1 < NS) __builtin_amdgcn_s_wait_tensorcnt(2);
            else            __builtin_amdgcn_s_wait_tensorcnt(0);
        }
        __syncthreads();                       // slab i resident for all waves

        const h16* Ab = &Alds[i & 1][0];
        const h16* Bb = &Blds[i & 1][0];
#pragma unroll
        for (int ks = 0; ks < 2; ++ks) {
            v16h af = ld_afrag(Ab + (wave * 16 + nlane) * STR + ks * 32, halfsel);
#pragma unroll
            for (int j = 0; j < 8; ++j) {
                v16h bf = ld_bfrag(Bb + (j * 16 + nlane) * STR + ks * 32, halfsel);
                acc[j] = wmma16(af, bf, acc[j]);
            }
        }
        __syncthreads();                       // all waves done with buffer i&1
        if (wave == 0 && (i + 2) < NS) {       // refill it with slab i+2
            tdm_load_2d_64(lds_offset_of(&Alds[i & 1][0]), aTile + (i + 2) * KS,
                           128, K, M);
            tdm_load_2d_64(lds_offset_of(&Blds[i & 1][0]), bTile + (i + 2) * KS,
                           128, K, N);
        }
    }

#pragma unroll
    for (int j = 0; j < 8; ++j)
#pragma unroll
        for (int r = 0; r < 8; ++r) {
            size_t gm = (size_t)(m0 + wave * 16 + mbase + r);
            size_t gn = (size_t)(n0 + j * 16 + nlane);
            if (STORE_F32) ((float*)Cout)[gm * N + gn] = acc[j][r];
            else           ((h16*)Cout)[gm * N + gn]   = (h16)acc[j][r];
        }
}

// ---------------------------------------------------------------------------
// RoPE + layout transform: y[B,S,H*HD] (f16) -> out[B,H,S,HD] (f16)
// Folds softmax scale into Q. ln(10000) = 9.210340371976184
// ---------------------------------------------------------------------------
__global__ __launch_bounds__(256) void rope_transpose(const h16* __restrict__ y,
                                                      h16* __restrict__ out,
                                                      int applyRope, float scale) {
    int idx = blockIdx.x * 256 + threadIdx.x;          // B*S*H*64 pairs
    int j = idx & 63;
    int h = (idx >> 6) & 31;
    int s = (idx >> 11) & 2047;
    int b = idx >> 22;
    const h16* src = y + (size_t)(b * 2048 + s) * 4096 + h * 128;
    float v1 = (float)src[j], v2 = (float)src[j + 64];
    float o1 = v1, o2 = v2;
    if (applyRope) {
        float invf = __expf(-(float)(2 * j) * (9.210340371976184f / 128.0f));
        float ang = (float)s * invf, sn, cs;
        __sincosf(ang, &sn, &cs);
        o1 = v1 * cs - v2 * sn;
        o2 = v2 * cs + v1 * sn;
    }
    h16* dst = out + ((size_t)(b * 32 + h) * 2048 + s) * 128;
    dst[j]      = (h16)(o1 * scale);
    dst[j + 64] = (h16)(o2 * scale);
}

// ---------------------------------------------------------------------------
// Causal flash attention. Grid: (B*H, S/128). Block = 256 = 8 waves.
// Q/K/V in [B,H,S,HD] f16 (scale pre-folded into Q). ctx out [B,S,H*HD] f16.
// ---------------------------------------------------------------------------
__global__ __launch_bounds__(256) void flash_attn(const h16* __restrict__ Q,
                                                  const h16* __restrict__ Km,
                                                  const h16* __restrict__ Vm,
                                                  h16* __restrict__ ctx) {
    constexpr int S = 2048, HD = 128, D = 4096;
    constexpr int KSTR = 136;   // 128 + 8
    constexpr int VSTR = 40;    // 32 + 8  (V stored transposed [hd][key])
    constexpr int PSTR = 40;
    __shared__ h16 Klds[32 * KSTR];
    __shared__ h16 Vlds[128 * VSTR];
    __shared__ h16 Plds[8 * 16 * PSTR];

    const int t = threadIdx.x, w = t >> 5, lane = t & 31;
    const int nlane = lane & 15, mbase = (lane >> 4) * 8, halfsel = lane >> 4;
    const int bh = blockIdx.x, q0 = blockIdx.y * 128;
    const h16* Qb = Q  + (size_t)bh * S * HD;
    const h16* Kb = Km + (size_t)bh * S * HD;
    const h16* Vb = Vm + (size_t)bh * S * HD;

    // Q fragments resident for the whole key loop (16 rows per wave, HD=128)
    v16h qf[4];
    {
        const h16* qrow = Qb + (size_t)(q0 + w * 16 + nlane) * HD;
#pragma unroll
        for (int tt = 0; tt < 4; ++tt) qf[tt] = ld_afrag(qrow + tt * 32, halfsel);
    }

    float mi[8], li[8];
    v8f o[8];
#pragma unroll
    for (int r = 0; r < 8; ++r) { mi[r] = -1e30f; li[r] = 0.0f; }
#pragma unroll
    for (int j = 0; j < 8; ++j)
#pragma unroll
        for (int r = 0; r < 8; ++r) o[j][r] = 0.0f;

    const int myq = q0 + w * 16;
    const int kend = q0 + 128;                 // causal upper bound

    for (int kb = 0; kb < kend; kb += 32) {
        __syncthreads();
        {   // stage K (natural) + V (transposed) tiles into LDS
            int key = t >> 3, c = (t & 7) * 16;
            const h16* gk = Kb + (size_t)(kb + key) * HD + c;
            *(v8h*)&Klds[key * KSTR + c]     = *(const v8h*)gk;
            *(v8h*)&Klds[key * KSTR + c + 8] = *(const v8h*)(gk + 8);
            const h16* gv = Vb + (size_t)(kb + key) * HD + c;
            v8h a0 = *(const v8h*)gv, a1 = *(const v8h*)(gv + 8);
#pragma unroll
            for (int i = 0; i < 8; ++i) {
                Vlds[(c + i) * VSTR + key]     = a0[i];
                Vlds[(c + 8 + i) * VSTR + key] = a1[i];
            }
        }
        __syncthreads();

        // S = Q @ K^T : 16 queries x 32 keys (two 16x16 tiles)
        v8f s0, s1;
#pragma unroll
        for (int i = 0; i < 8; ++i) { s0[i] = 0.0f; s1[i] = 0.0f; }
#pragma unroll
        for (int tt = 0; tt < 4; ++tt) {
            v16h k0f = ld_bfrag(&Klds[nlane * KSTR + tt * 32], halfsel);
            v16h k1f = ld_bfrag(&Klds[(16 + nlane) * KSTR + tt * 32], halfsel);
            s0 = wmma16(qf[tt], k0f, s0);
            s1 = wmma16(qf[tt], k1f, s1);
        }

        // causal mask (C layout: col = nlane, row = mbase+r)
        const int key0 = kb + nlane, key1 = kb + 16 + nlane;
#pragma unroll
        for (int r = 0; r < 8; ++r) {
            int qq = myq + mbase + r;
            if (key0 > qq) s0[r] = -1e30f;
            if (key1 > qq) s1[r] = -1e30f;
        }

        // online softmax: row reductions across the 16 lanes of each half-wave
        float alpha[8];
#pragma unroll
        for (int r = 0; r < 8; ++r) {
            float mx = fmaxf(s0[r], s1[r]);
            mx = fmaxf(mx, __shfl_xor(mx, 1));
            mx = fmaxf(mx, __shfl_xor(mx, 2));
            mx = fmaxf(mx, __shfl_xor(mx, 4));
            mx = fmaxf(mx, __shfl_xor(mx, 8));
            float mnew = fmaxf(mi[r], mx);
            alpha[r] = __expf(mi[r] - mnew);
            float p0 = __expf(s0[r] - mnew);
            float p1 = __expf(s1[r] - mnew);
            float ls = p0 + p1;
            ls += __shfl_xor(ls, 1);
            ls += __shfl_xor(ls, 2);
            ls += __shfl_xor(ls, 4);
            ls += __shfl_xor(ls, 8);
            li[r] = li[r] * alpha[r] + ls;
            mi[r] = mnew;
            s0[r] = p0; s1[r] = p1;
        }
#pragma unroll
        for (int j = 0; j < 8; ++j)
#pragma unroll
            for (int r = 0; r < 8; ++r) o[j][r] *= alpha[r];

        // P: C-layout -> A-layout via wave-private LDS patch
        h16* pw = &Plds[w * 16 * PSTR];
#pragma unroll
        for (int r = 0; r < 8; ++r) {
            pw[(mbase + r) * PSTR + nlane]      = (h16)s0[r];
            pw[(mbase + r) * PSTR + 16 + nlane] = (h16)s1[r];
        }
        // wave-local RAW through LDS: CDNA5 split DS counter, no barrier needed
        asm volatile("s_wait_dscnt 0" ::: "memory");
        v16h pf = ld_afrag(pw + nlane * PSTR, halfsel);

        // O += P @ V (V transposed in LDS -> contiguous B-fragment reads)
#pragma unroll
        for (int j = 0; j < 8; ++j) {
            v16h vf = ld_bfrag(&Vlds[(j * 16 + nlane) * VSTR], halfsel);
            o[j] = wmma16(pf, vf, o[j]);
        }
    }

    // normalize and store ctx in [B,S,H*HD] (row-major for the final GEMM)
    const int b = bh >> 5, h = bh & 31;
#pragma unroll
    for (int r = 0; r < 8; ++r) {
        float inv = 1.0f / li[r];
        size_t row = (size_t)(b * S + q0 + w * 16 + mbase + r);
        h16* dst = ctx + row * D + h * 128;
#pragma unroll
        for (int j = 0; j < 8; ++j) dst[j * 16 + nlane] = (h16)(o[j][r] * inv);
    }
}

// ---------------------------------------------------------------------------
// Host-side launch
// ---------------------------------------------------------------------------
extern "C" void kernel_launch(void* const* d_in, const int* in_sizes, int n_in,
                              void* d_out, int out_size, void* d_ws, size_t ws_size,
                              hipStream_t stream) {
    (void)in_sizes; (void)n_in; (void)out_size; (void)ws_size;
    constexpr int Bn = 2, S = 2048, D = 4096, H = 32;
    constexpr int M = Bn * S;                       // 4096
    constexpr size_t NE = (size_t)M * D;            // 16,777,216 elements
    constexpr size_t HB = NE * sizeof(h16);         // 32 MiB per f16 matrix

    const float* x  = (const float*)d_in[0];
    // d_in[1] = mask (causal tril, handled analytically in flash_attn)
    const float* Wq = (const float*)d_in[2];
    const float* Wk = (const float*)d_in[3];
    const float* Wv = (const float*)d_in[4];
    const float* Wo = (const float*)d_in[5];

    char* ws = (char*)d_ws;
    h16* xh  = (h16*)(ws + 0 * HB);
    h16* Wh0 = (h16*)(ws + 1 * HB);
    h16* Wh1 = (h16*)(ws + 2 * HB);
    h16* Wh2 = (h16*)(ws + 3 * HB);
    h16* Wh3 = (h16*)(ws + 4 * HB);
    h16* Qr  = (h16*)(ws + 5 * HB);
    h16* Kr  = (h16*)(ws + 6 * HB);
    h16* Vr  = (h16*)(ws + 7 * HB);
    h16* tmp = (h16*)(ws + 8 * HB);                 // GEMM scratch, later ctx

    const int cvtGrid = (int)(NE / (256 * 4));
    cvt_f32_f16<<<cvtGrid, 256, 0, stream>>>(x,  xh,  (int)NE);
    cvt_f32_f16<<<cvtGrid, 256, 0, stream>>>(Wq, Wh0, (int)NE);
    cvt_f32_f16<<<cvtGrid, 256, 0, stream>>>(Wk, Wh1, (int)NE);
    cvt_f32_f16<<<cvtGrid, 256, 0, stream>>>(Wv, Wh2, (int)NE);
    cvt_f32_f16<<<cvtGrid, 256, 0, stream>>>(Wo, Wh3, (int)NE);

    dim3 gGrid(M / 128, D / 128), gBlock(256);
    const int ropeGrid = Bn * S * H * 64 / 256;
    const float qscale = 0.08838834764831845f;      // 1/sqrt(128)

    wmma_gemm_tdm<false><<<gGrid, gBlock, 0, stream>>>(xh, Wh0, tmp, M, D, D);
    rope_transpose<<<ropeGrid, 256, 0, stream>>>(tmp, Qr, 1, qscale);
    wmma_gemm_tdm<false><<<gGrid, gBlock, 0, stream>>>(xh, Wh1, tmp, M, D, D);
    rope_transpose<<<ropeGrid, 256, 0, stream>>>(tmp, Kr, 1, 1.0f);
    wmma_gemm_tdm<false><<<gGrid, gBlock, 0, stream>>>(xh, Wh2, tmp, M, D, D);
    rope_transpose<<<ropeGrid, 256, 0, stream>>>(tmp, Vr, 0, 1.0f);

    dim3 fGrid(Bn * H, S / 128);
    flash_attn<<<fGrid, gBlock, 0, stream>>>(Qr, Kr, Vr, tmp);

    wmma_gemm_tdm<true><<<gGrid, gBlock, 0, stream>>>(tmp, Wh3, d_out, M, D, D);
}
